// DAWN_9646496547060
// MI455X (gfx1250) — compile-verified, matrix operand order checked by reading, not run on hardware
//
#include <hip/hip_runtime.h>
#include <hip/hip_bf16.h>
#include <math.h>

// ---------------------------------------------------------------------------
// P=2, B=4, S=1024, D=1024, R=16, DS=64, H=16, N_QK=N_V=32, N_RQK=N_RV=16.
// ---------------------------------------------------------------------------
#define TOKENS 4096
#define DMODEL 1024
#define DHEAD  64
#define SEQ    1024
#define BATCH  4

typedef unsigned short u16;
typedef __attribute__((ext_vector_type(16))) __bf16        v16bf;
typedef __attribute__((ext_vector_type(8)))  float         v8f;
typedef __attribute__((ext_vector_type(8)))  unsigned int  v8u;
typedef __attribute__((ext_vector_type(4)))  unsigned int  u32x4;
typedef __attribute__((ext_vector_type(8)))  int           i32x8;
typedef __attribute__((ext_vector_type(4)))  int           i32x4;

#if defined(__gfx1250__) && __has_builtin(__builtin_amdgcn_tensor_load_to_lds)
#define HAVE_TDM 1
#else
#define HAVE_TDM 0
#endif

__device__ __forceinline__ u16 f2bf(float f) {
    unsigned u = __builtin_bit_cast(unsigned, f);
    return (u16)((u + 0x7FFFu + ((u >> 16) & 1u)) >> 16);
}
__device__ __forceinline__ float bf2f(u16 h) {
    return __builtin_bit_cast(float, (unsigned)h << 16);
}
__device__ __forceinline__ unsigned pack2(float a, float b) {
    return (unsigned)f2bf(a) | ((unsigned)f2bf(b) << 16);
}

// 16-bf16 fragment from a K-contiguous row: elements 0..7 = K half*8..+7,
// elements 8..15 = K 16+half*8..+7  ->  two b128 loads.
__device__ __forceinline__ v16bf make_frag(const u16* row, int half) {
    uint4 a = *(const uint4*)(row + half * 8);
    uint4 b = *(const uint4*)(row + 16 + half * 8);
    v8u u;
    u[0] = a.x; u[1] = a.y; u[2] = a.z; u[3] = a.w;
    u[4] = b.x; u[5] = b.y; u[6] = b.z; u[7] = b.w;
    return __builtin_bit_cast(v16bf, u);
}

__device__ __forceinline__ float redmax16(float v) {
#pragma unroll
    for (int m = 1; m < 16; m <<= 1) v = fmaxf(v, __shfl_xor(v, m));
    return v;
}
__device__ __forceinline__ float redsum16(float v) {
#pragma unroll
    for (int m = 1; m < 16; m <<= 1) v += __shfl_xor(v, m);
    return v;
}

#if HAVE_TDM
// TDM 2D bf16 tile -> LDS, with LDS row padding (descriptor per CDNA5 ISA).
// tileW elems/row, tileH rows, global stride strideElems, pad after every
// (2<<padIntervalCode) dwords by (padAmountCode+1) dwords.
__device__ __forceinline__ void tdm_load_2d_bf16(void* lds, const void* g,
                                                 unsigned tileW, unsigned tileH,
                                                 unsigned strideElems,
                                                 unsigned padIntervalCode,
                                                 unsigned padAmountCode) {
    unsigned long long ga = (unsigned long long)(size_t)g;
    u32x4 g0;
    g0[0] = 1u;                                             // count=1 (valid)
    g0[1] = (unsigned)(size_t)lds;                          // lds_addr
    g0[2] = (unsigned)ga;                                   // global_addr[31:0]
    g0[3] = (unsigned)((ga >> 32) & 0x01FFFFFFu) | (2u << 30); // addr[56:32], type=2
    i32x8 g1;
    g1[0] = (1 << 16) | (1 << 20) | (int)(padIntervalCode << 22) | (int)(padAmountCode << 25);
    g1[1] = (int)0xFFFF0000;                 // tensor_dim0[15:0]=0xFFFF (huge)
    g1[2] = (int)0xFFFF7FFF;                 // tensor_dim0[31:16], tensor_dim1[15:0]
    g1[3] = (int)((tileW << 16) | 0x7FFF);   // tensor_dim1[31:16], tile_dim0
    g1[4] = (int)(tileH & 0xFFFF);           // tile_dim1, tile_dim2=0
    g1[5] = (int)strideElems;                // tensor_dim0_stride[31:0]
    g1[6] = 0;                               // stride hi, dim1_stride lo
    g1[7] = 0;
    i32x4 z4 = {0, 0, 0, 0};
    i32x8 z8 = {0, 0, 0, 0, 0, 0, 0, 0};
    __builtin_amdgcn_tensor_load_to_lds(g0, g1, z4, z4, z8, 0);
}
#endif

// ---------------------------------------------------------------------------
// Small setup kernels
// ---------------------------------------------------------------------------
__global__ void k_zero(float* p, int n) {
    int i = blockIdx.x * blockDim.x + threadIdx.x;
    if (i < n) p[i] = 0.f;
}

// f32 -> bf16 bulk convert (n multiple of 4)
__global__ __launch_bounds__(256) void k_cvt(const float* __restrict__ s,
                                             u16* __restrict__ d, int n4) {
    int i = blockIdx.x * 256 + threadIdx.x;
    if (i >= n4) return;
    float4 v = ((const float4*)s)[i];
    uint2 o; o.x = pack2(v.x, v.y); o.y = pack2(v.z, v.w);
    ((uint2*)d)[i] = o;
}

// E[n][t] = sum_j emb[n][j] * W[j][t]
__global__ __launch_bounds__(1024) void k_emb(const float* fqk_emb, const float* fv_emb,
                                              const float* W_rqk, const float* W_rv,
                                              float* Eqk, float* Ev) {
    int tid = threadIdx.x;
    int which = tid >> 9, rem = tid & 511;
    int n = rem >> 4, t = rem & 15;
    const float* emb = which ? fv_emb : fqk_emb;
    const float* W   = which ? W_rv   : W_rqk;
    float acc = 0.f;
    for (int j = 0; j < 64; j++) acc += emb[n * 64 + j] * W[j * 16 + t];
    (which ? Ev : Eqk)[n * 16 + t] = acc;
}

// WfpT[c][d] = f_neurons[c>>4][d][c&15], bf16, N(=c) x K(=d) row-major
__global__ __launch_bounds__(256) void k_repackF(const float* __restrict__ f,
                                                 u16* __restrict__ o) {
    int idx = blockIdx.x * 256 + threadIdx.x;   // 1024*1024
    int c = idx >> 10, d = idx & 1023;
    o[idx] = f2bf(f[(size_t)(c >> 4) * (DMODEL * 16) + d * 16 + (c & 15)]);
}

// Rt[bank][d][j] = r_neurons[bank*256+j][d], bf16 N(=d) x K(=j) row-major
__global__ __launch_bounds__(256) void k_repackR(const float* __restrict__ r,
                                                 u16* __restrict__ oqk,
                                                 u16* __restrict__ ov) {
    int idx = blockIdx.x * 256 + threadIdx.x;   // 2 * 1024 * 256
    int bank = idx >> 18, rem = idx & 262143;
    int d = rem >> 8, j = rem & 255;
    (bank ? ov : oqk)[rem] = f2bf(r[(size_t)(bank * 256 + j) * DMODEL + d]);
}

// ---------------------------------------------------------------------------
// bf16 WMMA GEMM: C(MxN) = A(MxK,bf16 row-major) * B(NxK,bf16 row-major)^T
// Tiles 128(M) x 64(N) x 64(K); LDS rows padded to 72 elems (conflict-free).
// TDM double-buffered pipeline: tile i+1 streams in while tile i computes.
// ---------------------------------------------------------------------------
__global__ __launch_bounds__(256) void k_gemm(const u16* __restrict__ A,
                                              const u16* __restrict__ B,
                                              void* __restrict__ Cp,
                                              int M, int N, int K, int outBf16) {
    __shared__ __align__(16) u16 lA[2][128 * 72];
    __shared__ __align__(16) u16 lB[2][64 * 72];

    int tid = threadIdx.x, wave = tid >> 5, lane = tid & 31;
    int half = lane >> 4, ln = lane & 15;
    int n0 = blockIdx.x * 64, m0 = blockIdx.y * 128;

    v8f acc0 = {0,0,0,0,0,0,0,0}, acc1 = {0,0,0,0,0,0,0,0};
    v8f acc2 = {0,0,0,0,0,0,0,0}, acc3 = {0,0,0,0,0,0,0,0};

    auto compute = [&](const u16* pA, const u16* pB) {
        const u16* ra = pA + (wave * 16 + ln) * 72;
        v16bf a0  = make_frag(ra, half);
        v16bf a1  = make_frag(ra + 32, half);
        v16bf b00 = make_frag(pB + (0  + ln) * 72,      half);
        v16bf b01 = make_frag(pB + (0  + ln) * 72 + 32, half);
        v16bf b10 = make_frag(pB + (16 + ln) * 72,      half);
        v16bf b11 = make_frag(pB + (16 + ln) * 72 + 32, half);
        v16bf b20 = make_frag(pB + (32 + ln) * 72,      half);
        v16bf b21 = make_frag(pB + (32 + ln) * 72 + 32, half);
        v16bf b30 = make_frag(pB + (48 + ln) * 72,      half);
        v16bf b31 = make_frag(pB + (48 + ln) * 72 + 32, half);
        acc0 = __builtin_amdgcn_wmma_f32_16x16x32_bf16(false, a0, false, b00, (short)0, acc0, false, false);
        acc0 = __builtin_amdgcn_wmma_f32_16x16x32_bf16(false, a1, false, b01, (short)0, acc0, false, false);
        acc1 = __builtin_amdgcn_wmma_f32_16x16x32_bf16(false, a0, false, b10, (short)0, acc1, false, false);
        acc1 = __builtin_amdgcn_wmma_f32_16x16x32_bf16(false, a1, false, b11, (short)0, acc1, false, false);
        acc2 = __builtin_amdgcn_wmma_f32_16x16x32_bf16(false, a0, false, b20, (short)0, acc2, false, false);
        acc2 = __builtin_amdgcn_wmma_f32_16x16x32_bf16(false, a1, false, b21, (short)0, acc2, false, false);
        acc3 = __builtin_amdgcn_wmma_f32_16x16x32_bf16(false, a0, false, b30, (short)0, acc3, false, false);
        acc3 = __builtin_amdgcn_wmma_f32_16x16x32_bf16(false, a1, false, b31, (short)0, acc3, false, false);
    };

#if HAVE_TDM
    const int nIter = K >> 6;
    if (wave == 0) {
        tdm_load_2d_bf16(lA[0], A + (size_t)m0 * K, 64, 128, K, 4, 3);
        tdm_load_2d_bf16(lB[0], B + (size_t)n0 * K, 64, 64,  K, 4, 3);
        __builtin_amdgcn_s_wait_tensorcnt(0);
    }
    __syncthreads();
    for (int it = 0; it < nIter; ++it) {
        const int cur = it & 1, nxt = cur ^ 1;
        if (wave == 0 && it + 1 < nIter) {
            int k0 = (it + 1) << 6;
            tdm_load_2d_bf16(lA[nxt], A + (size_t)m0 * K + k0, 64, 128, K, 4, 3);
            tdm_load_2d_bf16(lB[nxt], B + (size_t)n0 * K + k0, 64, 64,  K, 4, 3);
        }
        compute(lA[cur], lB[cur]);
        __syncthreads();                               // all done reading cur
        if (wave == 0 && it + 1 < nIter)
            __builtin_amdgcn_s_wait_tensorcnt(0);      // next tile resident
        __syncthreads();
    }
#else
    for (int k0 = 0; k0 < K; k0 += 64) {
        __syncthreads();
#pragma unroll
        for (int i = 0; i < 4; i++) {       // A: 128 rows x 8 uint4
            int idx = i * 256 + tid, r = idx >> 3, q = idx & 7;
            *(uint4*)(lA[0] + r * 72 + q * 8) =
                *(const uint4*)(A + (size_t)(m0 + r) * K + k0 + q * 8);
        }
#pragma unroll
        for (int i = 0; i < 2; i++) {       // B: 64 rows x 8 uint4
            int idx = i * 256 + tid, r = idx >> 3, q = idx & 7;
            *(uint4*)(lB[0] + r * 72 + q * 8) =
                *(const uint4*)(B + (size_t)(n0 + r) * K + k0 + q * 8);
        }
        __syncthreads();
        compute(lA[0], lB[0]);
    }
#endif

    if (outBf16) {
        u16* C = (u16*)Cp;
#pragma unroll
        for (int r = 0; r < 8; r++) {
            size_t row = (size_t)(m0 + wave * 16 + half * 8 + r) * N + n0;
            C[row + 0  + ln] = f2bf(acc0[r]);
            C[row + 16 + ln] = f2bf(acc1[r]);
            C[row + 32 + ln] = f2bf(acc2[r]);
            C[row + 48 + ln] = f2bf(acc3[r]);
        }
    } else {
        float* C = (float*)Cp;
#pragma unroll
        for (int r = 0; r < 8; r++) {
            size_t row = (size_t)(m0 + wave * 16 + half * 8 + r) * N + n0;
            C[row + 0  + ln] = acc0[r];
            C[row + 16 + ln] = acc1[r];
            C[row + 32 + ln] = acc2[r];
            C[row + 48 + ln] = acc3[r];
        }
    }
}

// ---------------------------------------------------------------------------
// Per-token fusion (wave per token; lane = (p<<4)|t).
// ---------------------------------------------------------------------------
__global__ __launch_bounds__(256) void k_token(const u16* __restrict__ Hall,
                                               const float* __restrict__ fqk_Q,
                                               const float* __restrict__ fqk_K,
                                               const float* __restrict__ fv_w,
                                               const float* __restrict__ gQ,
                                               const float* __restrict__ gK,
                                               const float* __restrict__ gV,
                                               const float* __restrict__ Eqk,
                                               const float* __restrict__ Ev,
                                               const float* __restrict__ W_rqk,
                                               const float* __restrict__ W_rv,
                                               float* __restrict__ auxsum,
                                               u16* __restrict__ Tq,
                                               u16* __restrict__ Tk,
                                               u16* __restrict__ Tv) {
    int wave = threadIdx.x >> 5, lane = threadIdx.x & 31;
    int tok = blockIdx.x * 8 + wave;
    int b = tok >> 10, s = tok & 1023;
    int p = lane >> 4, t = lane & 15;
    const u16* allh = Hall + (size_t)tok * 1024;     // [n(64)][r(16)] bf16

    const float* wsrc[3] = {fqk_Q, fqk_K, fv_w};
    const float* gsrc[3] = {gQ, gK, gV};
    const float* Esrc[3] = {Eqk, Eqk, Ev};
    const float* Wt[3]   = {W_rqk + 64 * 16, W_rqk + 64 * 16, W_rv + 64 * 16};
    u16*         Tdst[3] = {Tq, Tk, Tv};
    const int    nOff[3] = {0, 0, 32};

    for (int st = 0; st < 3; st++) {
        size_t wbase = ((size_t)(p * BATCH + b) * SEQ + s) * 32;
        float w[32];
#pragma unroll
        for (int n = 0; n < 32; n++) w[n] = wsrc[st][wbase + n];

        float h = 0.f;
#pragma unroll
        for (int n = 0; n < 32; n++) h += bf2f(allh[(nOff[st] + n) * 16 + t]) * w[n];

        float lg = 0.f;
#pragma unroll
        for (int n = 0; n < 32; n++) lg += w[n] * Esrc[st][n * 16 + t];
#pragma unroll
        for (int r = 0; r < 16; r++)
            lg += __shfl(h, (lane & 16) | r) * Wt[st][r * 16 + t];

        float mx = redmax16(lg);
        float e = __expf(lg - mx);
        float sm = redsum16(e);
        float rw = (e / sm) * gsrc[st][(size_t)(p * BATCH + b) * SEQ + s];

        atomicAdd(&auxsum[st * 32 + p * 16 + t], rw);

#pragma unroll
        for (int r = 0; r < 16; r++) {
            float c  = __shfl(h, (lane & 16) | r) * rw;
            float cf = c + __shfl_xor(c, 16);
            if (p == 0) Tdst[st][(size_t)tok * 256 + t * 16 + r] = f2bf(cf);
        }
    }
}

// qscale[tok] from ||Q[tok,:]|| (full D)
__global__ __launch_bounds__(256) void k_qnorm(const u16* __restrict__ Q,
                                               float* __restrict__ qs) {
    int wave = threadIdx.x >> 5, lane = threadIdx.x & 31;
    int tok = blockIdx.x * 8 + wave;
    const u16* q = Q + (size_t)tok * DMODEL;
    float s = 0.f;
    for (int i = lane; i < DMODEL; i += 32) { float v = bf2f(q[i]); s += v * v; }
#pragma unroll
    for (int m = 16; m >= 1; m >>= 1) s += __shfl_xor(s, m);
    if (lane == 0) { float n = sqrtf(s); qs[tok] = (n > 1e-6f) ? 1.f : n * 1e-6f; }
}

// V[b][s][h*64+d] -> VT[(b*16+h)*64+d][s]  (per-head transpose, tiled)
__global__ __launch_bounds__(256) void k_vt(const u16* __restrict__ V,
                                            u16* __restrict__ VT) {
    __shared__ __align__(16) u16 lT[64 * 72];
    int bh = blockIdx.x, s0 = blockIdx.y * 64;
    int b = bh >> 4, h = bh & 15;
    int tid = threadIdx.x;
#pragma unroll
    for (int i = 0; i < 2; i++) {
        int idx = i * 256 + tid, s = idx >> 3, q = idx & 7;
        *(uint4*)(lT + s * 72 + q * 8) =
            *(const uint4*)(V + (size_t)(b * SEQ + s0 + s) * DMODEL + h * 64 + q * 8);
    }
    __syncthreads();
#pragma unroll
    for (int i = 0; i < 16; i++) {
        int idx = i * 256 + tid, d = idx >> 6, s = idx & 63;
        VT[((size_t)bh * 64 + d) * SEQ + s0 + s] = lT[s * 72 + d];
    }
}

// ---------------------------------------------------------------------------
// Flash attention, causal, dh=64. 8 waves x 16 q-rows per WG. K/V(T) tiles
// TDM double-buffered; scores scaled 1/8 post-WMMA; q-norm scale in epilogue.
// ---------------------------------------------------------------------------
__global__ __launch_bounds__(256) void k_attn(const u16* __restrict__ Q,
                                              const u16* __restrict__ Kk,
                                              const u16* __restrict__ VT,
                                              const float* __restrict__ qscale,
                                              u16* __restrict__ O) {
    __shared__ __align__(16) u16 lK[2][32 * 72];
    __shared__ __align__(16) u16 lV[2][64 * 40];
    __shared__ __align__(16) u16 lP[8 * 16 * 40];

    int tid = threadIdx.x, wave = tid >> 5, lane = tid & 31;
    int half = lane >> 4, ln = lane & 15;
    int bh = blockIdx.y; int b = bh >> 4, hh = bh & 15;
    int q0 = blockIdx.x * 128;
    int qrow = q0 + wave * 16 + ln;

    const u16* qp = Q + ((size_t)b * SEQ + qrow) * DMODEL + hh * DHEAD;
    v16bf qf0 = make_frag(qp, half);
    v16bf qf1 = make_frag(qp + 32, half);

    float mrow[8], lrow[8];
#pragma unroll
    for (int r = 0; r < 8; r++) { mrow[r] = -1e30f; lrow[r] = 0.f; }
    v8f o0 = {0,0,0,0,0,0,0,0}, o1 = {0,0,0,0,0,0,0,0};
    v8f o2 = {0,0,0,0,0,0,0,0}, o3 = {0,0,0,0,0,0,0,0};

    auto attn_block = [&](int kBase, const u16* pK, const u16* pV) {
        v8f sA = {0,0,0,0,0,0,0,0}, sB = {0,0,0,0,0,0,0,0};
        sA = __builtin_amdgcn_wmma_f32_16x16x32_bf16(false, qf0, false, make_frag(pK + ln * 72,             half), (short)0, sA, false, false);
        sA = __builtin_amdgcn_wmma_f32_16x16x32_bf16(false, qf1, false, make_frag(pK + ln * 72 + 32,        half), (short)0, sA, false, false);
        sB = __builtin_amdgcn_wmma_f32_16x16x32_bf16(false, qf0, false, make_frag(pK + (16 + ln) * 72,      half), (short)0, sB, false, false);
        sB = __builtin_amdgcn_wmma_f32_16x16x32_bf16(false, qf1, false, make_frag(pK + (16 + ln) * 72 + 32, half), (short)0, sB, false, false);

        int kc0 = kBase + ln, kc1 = kBase + 16 + ln;
        float newm[8], psum[8];
#pragma unroll
        for (int r = 0; r < 8; r++) {
            int qr = q0 + wave * 16 + half * 8 + r;
            float a  = (kc0 <= qr) ? sA[r] * 0.125f : -1e9f;
            float bb = (kc1 <= qr) ? sB[r] * 0.125f : -1e9f;
            sA[r] = a; sB[r] = bb;
            newm[r] = fmaxf(mrow[r], redmax16(fmaxf(a, bb)));
        }
#pragma unroll
        for (int r = 0; r < 8; r++) {
            float ea = __expf(sA[r] - newm[r]);
            float eb = __expf(sB[r] - newm[r]);
            sA[r] = ea; sB[r] = eb;
            psum[r] = redsum16(ea + eb);
        }
        u16* pb = lP + wave * (16 * 40);
#pragma unroll
        for (int r = 0; r < 8; r++) {
            float corr = __expf(mrow[r] - newm[r]);
            lrow[r] = lrow[r] * corr + psum[r];
            mrow[r] = newm[r];
            o0[r] *= corr; o1[r] *= corr; o2[r] *= corr; o3[r] *= corr;
            pb[(half * 8 + r) * 40 + ln]      = f2bf(sA[r]);
            pb[(half * 8 + r) * 40 + 16 + ln] = f2bf(sB[r]);
        }
        v16bf pf = make_frag(pb + ln * 40, half);
        o0 = __builtin_amdgcn_wmma_f32_16x16x32_bf16(false, pf, false, make_frag(pV + (0  + ln) * 40, half), (short)0, o0, false, false);
        o1 = __builtin_amdgcn_wmma_f32_16x16x32_bf16(false, pf, false, make_frag(pV + (16 + ln) * 40, half), (short)0, o1, false, false);
        o2 = __builtin_amdgcn_wmma_f32_16x16x32_bf16(false, pf, false, make_frag(pV + (32 + ln) * 40, half), (short)0, o2, false, false);
        o3 = __builtin_amdgcn_wmma_f32_16x16x32_bf16(false, pf, false, make_frag(pV + (48 + ln) * 40, half), (short)0, o3, false, false);
    };

    const int nblk = q0 / 32 + 4;
#if HAVE_TDM
    if (wave == 0) {
        tdm_load_2d_bf16(lK[0], Kk + (size_t)b * SEQ * DMODEL + hh * DHEAD, 64, 32, DMODEL, 4, 3);
        tdm_load_2d_bf16(lV[0], VT + (size_t)bh * 64 * SEQ,                 32, 64, SEQ,    3, 3);
        __builtin_amdgcn_s_wait_tensorcnt(0);
    }
    __syncthreads();
    for (int kb = 0; kb < nblk; kb++) {
        const int cur = kb & 1, nxt = cur ^ 1;
        if (wave == 0 && kb + 1 < nblk) {
            int kn = (kb + 1) * 32;
            tdm_load_2d_bf16(lK[nxt], Kk + ((size_t)b * SEQ + kn) * DMODEL + hh * DHEAD, 64, 32, DMODEL, 4, 3);
            tdm_load_2d_bf16(lV[nxt], VT + (size_t)bh * 64 * SEQ + kn,                   32, 64, SEQ,    3, 3);
        }
        attn_block(kb * 32, lK[cur], lV[cur]);
        __syncthreads();
        if (wave == 0 && kb + 1 < nblk)
            __builtin_amdgcn_s_wait_tensorcnt(0);
        __syncthreads();
    }
#else
    for (int kb = 0; kb < nblk; kb++) {
        int kBase = kb * 32;
        __syncthreads();
        {   // lK: 32 rows x 8 uint4
            int r = tid >> 3, q = tid & 7;
            *(uint4*)(lK[0] + r * 72 + q * 8) =
                *(const uint4*)(Kk + ((size_t)b * SEQ + kBase + r) * DMODEL + hh * DHEAD + q * 8);
        }
        {   // lV: 64 rows x 4 uint4
            int r = tid >> 2, q = tid & 3;
            *(uint4*)(lV[0] + r * 40 + q * 8) =
                *(const uint4*)(VT + ((size_t)bh * 64 + r) * SEQ + kBase + q * 8);
        }
        __syncthreads();
        attn_block(kBase, lK[0], lV[0]);
    }
#endif

#pragma unroll
    for (int r = 0; r < 8; r++) {
        int s = q0 + wave * 16 + half * 8 + r;
        float f = qscale[b * SEQ + s] / lrow[r];
        u16* op = O + ((size_t)b * SEQ + s) * DMODEL + hh * DHEAD;
        op[0  + ln] = f2bf(o0[r] * f);
        op[16 + ln] = f2bf(o1[r] * f);
        op[32 + ln] = f2bf(o2[r] * f);
        op[48 + ln] = f2bf(o3[r] * f);
    }
}

__global__ void k_aux(const float* auxsum, float* out_aux) {
    float a = 0.f;
    for (int i = 0; i < 96; i++) { float m = auxsum[i] * (1.f / TOKENS); a += m * m; }
    out_aux[0] = 16.f * a;
}

// ---------------------------------------------------------------------------
// Host launcher
// ---------------------------------------------------------------------------
extern "C" void kernel_launch(void* const* d_in, const int* in_sizes, int n_in,
                              void* d_out, int out_size, void* d_ws, size_t ws_size,
                              hipStream_t stream) {
    const float* x       = (const float*)d_in[0];
    const float* fqk_Q   = (const float*)d_in[1];
    const float* fqk_K   = (const float*)d_in[2];
    const float* fv_w    = (const float*)d_in[3];
    const float* gate_Q  = (const float*)d_in[4];
    const float* gate_K  = (const float*)d_in[5];
    const float* gate_V  = (const float*)d_in[6];
    const float* f_neur  = (const float*)d_in[7];
    const float* r_neur  = (const float*)d_in[8];
    const float* fqk_emb = (const float*)d_in[9];
    const float* fv_emb  = (const float*)d_in[10];
    const float* W_rqk   = (const float*)d_in[11];
    const float* W_rv    = (const float*)d_in[12];
    const float* W_O     = (const float*)d_in[13];
    float* out = (float*)d_out;

    char* ws = (char*)d_ws;
    const size_t MB = 1024 * 1024;
    u16* XBF  = (u16*)(ws + 0);            // 4M elems (8 MB)
    u16* HALL = (u16*)(ws + 8  * MB);      // 4M elems; reused as AO
    u16* WFPT = (u16*)(ws + 16 * MB);      // 1M elems
    u16* RTQK = (u16*)(ws + 18 * MB);      // 256K elems
    u16* RTV  = (u16*)(ws + 19 * MB);      // 256K elems
    u16* WOBF = (u16*)(ws + 20 * MB);      // 1M elems
    u16* TQ   = (u16*)(ws + 22 * MB);      // 1M elems each
    u16* TK   = (u16*)(ws + 24 * MB);
    u16* TV   = (u16*)(ws + 26 * MB);
    u16* QB   = (u16*)(ws + 28 * MB);      // 4M elems each
    u16* KB   = (u16*)(ws + 36 * MB);
    u16* VB   = (u16*)(ws + 44 * MB);
    u16* VT   = (u16*)(ws + 52 * MB);
    float* QSC = (float*)(ws + 60 * MB);   // 4096
    float* AUX = QSC + 4096;               // 96
    float* EQK = AUX + 96;                 // 512
    float* EV  = EQK + 512;                // 512
    u16* AO = HALL;

    k_zero<<<1, 96, 0, stream>>>(AUX, 96);
    k_emb<<<1, 1024, 0, stream>>>(fqk_emb, fv_emb, W_rqk, W_rv, EQK, EV);
    k_cvt<<<4096, 256, 0, stream>>>(x, XBF, 1048576);         // x -> bf16
    k_cvt<<<1024, 256, 0, stream>>>(W_O, WOBF, 262144);       // W_O -> bf16 (NxK)
    k_repackF<<<4096, 256, 0, stream>>>(f_neur, WFPT);
    k_repackR<<<2048, 256, 0, stream>>>(r_neur, RTQK, RTV);

    k_gemm<<<dim3(16, 32), 256, 0, stream>>>(XBF, WFPT, HALL, TOKENS, 1024, 1024, 1);

    k_token<<<512, 256, 0, stream>>>(HALL, fqk_Q, fqk_K, fv_w, gate_Q, gate_K, gate_V,
                                     EQK, EV, W_rqk, W_rv, AUX, TQ, TK, TV);

    k_gemm<<<dim3(16, 32), 256, 0, stream>>>(TQ, RTQK, QB, TOKENS, 1024, 256, 1);
    k_gemm<<<dim3(16, 32), 256, 0, stream>>>(TK, RTQK, KB, TOKENS, 1024, 256, 1);
    k_gemm<<<dim3(16, 32), 256, 0, stream>>>(TV, RTV,  VB, TOKENS, 1024, 256, 1);

    k_qnorm<<<512, 256, 0, stream>>>(QB, QSC);
    k_vt<<<dim3(64, 16), 256, 0, stream>>>(VB, VT);
    k_attn<<<dim3(8, 64), 256, 0, stream>>>(QB, KB, VT, QSC, AO);

    k_gemm<<<dim3(16, 32), 256, 0, stream>>>(AO, WOBF, out, TOKENS, 1024, 1024, 0);

    k_aux<<<1, 1, 0, stream>>>(AUX, out + (size_t)TOKENS * DMODEL);
}
